// GCN_38929583571058
// MI455X (gfx1250) — matrix operand order repaired and checked by reading, not compiled, not run on hardware
//
#include <hip/hip_runtime.h>
#include <hip/hip_bf16.h>
#include <hip/hip_fp16.h>

// ---------- types ----------
typedef __attribute__((ext_vector_type(16))) _Float16 v16h;
typedef __attribute__((ext_vector_type(8)))  float    v8f;

#define F_HID 64
#define F_OUT 10

__device__ __forceinline__ void atomAddF(float* p, float v) {
  // hardware global_atomic_add_f32 (no CAS loop)
  unsafeAtomicAdd(p, v);
}

// ---------- degree / normalization ----------
__global__ void deg_init_kernel(float* __restrict__ deg, int n) {
  int i = blockIdx.x * blockDim.x + threadIdx.x;
  if (i < n) deg[i] = 1.0f;  // self loop
}

__global__ void deg_edge_kernel(const int* __restrict__ col, float* __restrict__ deg, int e) {
  int i = blockIdx.x * blockDim.x + threadIdx.x;
  if (i < e) atomAddF(&deg[col[i]], 1.0f);
}

__global__ void dinv_kernel(float* __restrict__ deg, int n) {
  int i = blockIdx.x * blockDim.x + threadIdx.x;
  if (i < n) {
    float d = deg[i];
    deg[i] = (d > 0.0f) ? rsqrtf(d) : 0.0f;
  }
}

__global__ void norm_kernel(const int* __restrict__ row, const int* __restrict__ col,
                            const float* __restrict__ dinv, float* __restrict__ norm, int e) {
  int i = blockIdx.x * blockDim.x + threadIdx.x;
  if (i < e) norm[i] = dinv[row[i]] * dinv[col[i]];
}

// ---------- WMMA GEMM: C[N x 64] = A[N x KDIM] @ W[KDIM x 64] ----------
// block = 256 threads = 8 waves; wave (wr,wc) computes a 16x16 tile;
// block covers 32 rows x 64 cols. W is staged once per block into LDS,
// converted to f16 and pre-swizzled into WMMA B-fragment order so each
// wave's B fragment is one contiguous 32-byte LDS read.
//
// LDS half index for element W[k][col]:
//   kk = k>>5, hi = (k>>4)&1, j = k&15, ct = col>>4, n = col&15
//   idx = ((((kk*2 + hi)*4 + ct)*16 + n) << 4) + j
template <int KDIM>
__global__ void gemm_wmma_kernel(const float* __restrict__ A, const float* __restrict__ W,
                                 float* __restrict__ C, int nrows) {
  __shared__ _Float16 wlds[KDIM * F_HID];

  // cooperative fill: global f32 -> LDS f16, fragment-swizzled
  for (int idx = threadIdx.x; idx < KDIM * F_HID; idx += blockDim.x) {
    int k   = idx >> 6;     // 0..KDIM-1
    int col = idx & 63;     // 0..63
    int kk = k >> 5;
    int hi = (k >> 4) & 1;
    int j  = k & 15;
    int ct = col >> 4;
    int n  = col & 15;
    int lidx = ((((kk * 2 + hi) * 4 + ct) * 16 + n) << 4) + j;
    wlds[lidx] = (_Float16)W[idx];
  }
  __syncthreads();

  const int lane = threadIdx.x & 31;
  const int wave = threadIdx.x >> 5;
  const int wr = wave >> 2;         // 0..1
  const int wc = wave & 3;          // 0..3
  const int row0 = blockIdx.x * 32 + wr * 16;
  const int col0 = wc * 16;
  const int m  = lane & 15;         // A: row within tile; B/C: col within tile
  const int hi = lane >> 4;         // half-wave select

  const int arow = row0 + m;
  const bool rowok = (arow < nrows);

  v8f acc = {};
#pragma unroll
  for (int kk = 0; kk < KDIM / 32; ++kk) {
    const int k0 = kk * 32;
    // A fragment (16x32 f16): lane holds A[m][hi*16 + j], j=0..15
    v16h a;
    if (rowok) {
      const float* ap = A + (size_t)arow * KDIM + k0 + hi * 16;
#pragma unroll
      for (int j = 0; j < 8; ++j) {
        a[j]     = (_Float16)ap[j];
        a[8 + j] = (_Float16)ap[8 + j];
      }
    } else {
#pragma unroll
      for (int j = 0; j < 16; ++j) a[j] = (_Float16)0.0f;
    }
    // B fragment (32x16 f16): lane holds W[k0 + hi*16 + j][col0 + m], j=0..15
    // contiguous 32B in LDS thanks to the swizzled fill
    const v16h b = *(const v16h*)&wlds[((((kk * 2 + hi) * 4 + wc) * 16 + m) << 4)];

    acc = __builtin_amdgcn_wmma_f32_16x16x32_f16(
        /*neg_a=*/false, a, /*neg_b=*/false, b,
        /*c_mod=*/(short)0, acc, /*reuse_a=*/false, /*reuse_b=*/false);
  }
  // C fragment: lane -> col (col0+m), VGPR r -> row (row0 + hi*8 + r)
#pragma unroll
  for (int r = 0; r < 8; ++r) {
    int rr = row0 + hi * 8 + r;
    if (rr < nrows) C[(size_t)rr * F_HID + col0 + m] = acc[r];
  }
}

// ---------- edge scatter: agg[row[e]] += t[col[e]] * norm[e] ----------
// one wave per edge; lane handles features lane and lane+32
__global__ void scatter_kernel(const int* __restrict__ row, const int* __restrict__ col,
                               const float* __restrict__ norm, const float* __restrict__ t,
                               float* __restrict__ agg, int e) {
  int gid  = blockIdx.x * blockDim.x + threadIdx.x;
  int edge = gid >> 5;
  int lane = threadIdx.x & 31;
  if (edge < e) {
    int r = row[edge];
    int c = col[edge];
    float nm = norm[edge];
    const float* src = t + (size_t)c * F_HID;
    float* dst = agg + (size_t)r * F_HID;
    atomAddF(dst + lane,      src[lane]      * nm);
    atomAddF(dst + lane + 32, src[lane + 32] * nm);
  }
}

// ---------- pointwise: x = relu(agg + t*dinv^2 + b); agg := x; accsum += x ----------
__global__ void pointwise_kernel(float* __restrict__ agg, const float* __restrict__ t,
                                 const float* __restrict__ dinv, const float* __restrict__ b,
                                 float* __restrict__ accsum, int n) {
  int idx = blockIdx.x * blockDim.x + threadIdx.x;
  if (idx < n * F_HID) {
    int i = idx >> 6;       // node
    int f = idx & 63;       // feature
    float di = dinv[i];
    float v = agg[idx] + t[idx] * di * di + b[f];
    float x = fmaxf(v, 0.0f);
    agg[idx] = x;
    accsum[idx] += x;
  }
}

// ---------- global mean pool (sum + count) ----------
__global__ void pool_kernel(const int* __restrict__ batch, const float* __restrict__ accsum,
                            float* __restrict__ pooled, float* __restrict__ cnt, int n) {
  int gid  = blockIdx.x * blockDim.x + threadIdx.x;
  int i    = gid >> 5;
  int lane = threadIdx.x & 31;
  if (i < n) {
    int g = batch[i];
    const float* src = accsum + (size_t)i * F_HID;
    float* dst = pooled + (size_t)g * F_HID;
    atomAddF(dst + lane,      src[lane]);
    atomAddF(dst + lane + 32, src[lane + 32]);
    if (lane == 0) atomAddF(&cnt[g], 1.0f);
  }
}

// ---------- head: softmax(pooled/(3*cnt) @ Wl + bl) ----------
__global__ void head_kernel(const float* __restrict__ pooled, const float* __restrict__ cnt,
                            const float* __restrict__ Wl, const float* __restrict__ bl,
                            float* __restrict__ out, int g_total) {
  int g = blockIdx.x * blockDim.x + threadIdx.x;
  if (g >= g_total) return;
  float inv = 1.0f / (3.0f * fmaxf(cnt[g], 1.0f));
  const float* p = pooled + (size_t)g * F_HID;
  float logit[F_OUT];
#pragma unroll
  for (int o = 0; o < F_OUT; ++o) logit[o] = bl[o];
  for (int f = 0; f < F_HID; ++f) {
    float pv = p[f] * inv;
#pragma unroll
    for (int o = 0; o < F_OUT; ++o) logit[o] += pv * Wl[f * F_OUT + o];
  }
  float mx = logit[0];
#pragma unroll
  for (int o = 1; o < F_OUT; ++o) mx = fmaxf(mx, logit[o]);
  float s = 0.0f;
#pragma unroll
  for (int o = 0; o < F_OUT; ++o) { logit[o] = expf(logit[o] - mx); s += logit[o]; }
  float is = 1.0f / s;
#pragma unroll
  for (int o = 0; o < F_OUT; ++o) out[(size_t)g * F_OUT + o] = logit[o] * is;
}

// ---------- launch ----------
extern "C" void kernel_launch(void* const* d_in, const int* in_sizes, int n_in,
                              void* d_out, int out_size, void* d_ws, size_t ws_size,
                              hipStream_t stream) {
  const float* features = (const float*)d_in[0];
  const int*   edge     = (const int*)d_in[1];
  const int*   batch    = (const int*)d_in[2];
  // d_in[3] = num_graphs (device scalar; G derived from out_size instead)
  const float* W1 = (const float*)d_in[4];
  const float* b1 = (const float*)d_in[5];
  const float* W2 = (const float*)d_in[6];
  const float* b2 = (const float*)d_in[7];
  const float* W3 = (const float*)d_in[8];
  const float* b3 = (const float*)d_in[9];
  const float* Wl = (const float*)d_in[10];
  const float* bl = (const float*)d_in[11];
  float* out = (float*)d_out;

  const int FIN = in_sizes[4] / F_HID;   // 128
  const int N   = in_sizes[0] / FIN;     // 100000
  const int E   = in_sizes[1] / 2;       // 3200000
  const int G   = out_size / F_OUT;      // 1000
  const int* erow = edge;        // destinations (scatter target)
  const int* ecol = edge + E;    // sources (message origin)

  // workspace carve-up (f32, 256B-aligned chunks)
  float* ws = (float*)d_ws;
  size_t off = 0;
  auto alloc = [&](size_t cnt_) {
    float* p = ws + off;
    off += (cnt_ + 63) & ~(size_t)63;
    return p;
  };
  float* dinv   = alloc((size_t)N);
  float* norm   = alloc((size_t)E);
  float* t      = alloc((size_t)N * F_HID);
  float* bufA   = alloc((size_t)N * F_HID);
  float* bufB   = alloc((size_t)N * F_HID);
  float* accsum = alloc((size_t)N * F_HID);
  float* pooled = alloc((size_t)G * F_HID);
  float* cnt    = alloc((size_t)G);
  (void)ws_size;

  const int T = 256;
  const size_t nfBytes = (size_t)N * F_HID * sizeof(float);
  const int nBlocks    = (N + T - 1) / T;
  const int eBlocks    = (E + T - 1) / T;
  const int gemmBlocks = (N + 31) / 32;
  const int edgeWaveBlocks = (int)(((size_t)E * 32 + T - 1) / T);
  const int nodeWaveBlocks = (int)(((size_t)N * 32 + T - 1) / T);
  const int nfBlocks       = (int)(((size_t)N * F_HID + T - 1) / T);
  const int gBlocks        = (G + T - 1) / T;

  // GCN normalization
  deg_init_kernel<<<nBlocks, T, 0, stream>>>(dinv, N);
  deg_edge_kernel<<<eBlocks, T, 0, stream>>>(ecol, dinv, E);
  dinv_kernel<<<nBlocks, T, 0, stream>>>(dinv, N);
  norm_kernel<<<eBlocks, T, 0, stream>>>(erow, ecol, dinv, norm, E);

  hipMemsetAsync(accsum, 0, nfBytes, stream);

  // layer 1: t = features @ W1 ; bufA = relu(scatter(t) + self + b1)
  gemm_wmma_kernel<128><<<gemmBlocks, T, 0, stream>>>(features, W1, t, N);
  hipMemsetAsync(bufA, 0, nfBytes, stream);
  scatter_kernel<<<edgeWaveBlocks, T, 0, stream>>>(erow, ecol, norm, t, bufA, E);
  pointwise_kernel<<<nfBlocks, T, 0, stream>>>(bufA, t, dinv, b1, accsum, N);

  // layer 2
  gemm_wmma_kernel<64><<<gemmBlocks, T, 0, stream>>>(bufA, W2, t, N);
  hipMemsetAsync(bufB, 0, nfBytes, stream);
  scatter_kernel<<<edgeWaveBlocks, T, 0, stream>>>(erow, ecol, norm, t, bufB, E);
  pointwise_kernel<<<nfBlocks, T, 0, stream>>>(bufB, t, dinv, b2, accsum, N);

  // layer 3 (reuse bufA; x1 no longer needed)
  gemm_wmma_kernel<64><<<gemmBlocks, T, 0, stream>>>(bufB, W3, t, N);
  hipMemsetAsync(bufA, 0, nfBytes, stream);
  scatter_kernel<<<edgeWaveBlocks, T, 0, stream>>>(erow, ecol, norm, t, bufA, E);
  pointwise_kernel<<<nfBlocks, T, 0, stream>>>(bufA, t, dinv, b3, accsum, N);

  // mean pool + classifier head
  hipMemsetAsync(pooled, 0, (size_t)G * F_HID * sizeof(float), stream);
  hipMemsetAsync(cnt, 0, (size_t)G * sizeof(float), stream);
  pool_kernel<<<nodeWaveBlocks, T, 0, stream>>>(batch, accsum, pooled, cnt, N);
  head_kernel<<<gBlocks, T, 0, stream>>>(pooled, cnt, Wl, bl, out, G);
}
